// RWKV_Tmix_x070_Mose_cxa079_38560216383806
// MI455X (gfx1250) — matrix operand order repaired and checked by reading, not compiled, not run on hardware
//
#include <hip/hip_runtime.h>
#include <math.h>

// ---------------------------------------------------------------------------
// Types for CDNA5 WMMA (gfx1250, wave32)
// ---------------------------------------------------------------------------
typedef __attribute__((ext_vector_type(16))) __bf16 v16bf;
typedef __attribute__((ext_vector_type(8)))  __bf16 bf16x8;
typedef __attribute__((ext_vector_type(8)))  float  v8f;

#define T_SEQ 2048
#define C_DIM 2048
#define HEADS 16
#define HDIM  128
#define KVH_DIM 512

// ---------------------------------------------------------------------------
// bf16 WMMA GEMM: C[M,N] = (A[M,K] * rowScale[m] * eltMul[m,k]) @ B[K,N]
// - A,B f32 row-major, converted to bf16 while staging to LDS
// - 128x128 block tile, K-step 32; 8 waves (4Mx2N), 32x64 patch per wave
// - double-buffered LDS, register-staged global loads (loads for tile k+1
//   in flight while the 8 v_wmma_f32_16x16x32_bf16 of tile k execute)
// - requires K%32==0, M%128==0, N%16==0 (true for every GEMM in this model)
// ---------------------------------------------------------------------------
#define BM 128
#define BN 128
#define BK 32
#define LDST 40   // LDS row stride in bf16 elems (80B rows -> 16B aligned chunks)

template <bool HAS_RS, bool HAS_EM>
__global__ __launch_bounds__(256) void gemm_bf16_wmma(
    const float* __restrict__ A, const float* __restrict__ B,
    float* __restrict__ C, int M, int N, int K,
    const float* __restrict__ rowScale, const float* __restrict__ eltMul)
{
    __shared__ __bf16 As[2][BM * LDST];   // [row][k]
    __shared__ __bf16 Bs[2][BN * LDST];   // [col][k]  (B tile stored transposed)

    const int tid   = threadIdx.x;
    const int lane  = tid & 31;
    const int wave  = tid >> 5;
    const int waveM = wave & 3;        // 0..3 -> 32-row strip
    const int waveN = wave >> 2;       // 0..1 -> 64-col strip
    const int lr    = lane & 15;
    const int lh    = lane >> 4;

    const int blockM = blockIdx.y * BM;
    const int blockN = blockIdx.x * BN;

    v8f acc[2][4];
#pragma unroll
    for (int mi = 0; mi < 2; ++mi)
#pragma unroll
        for (int ni = 0; ni < 4; ++ni)
            acc[mi][ni] = (v8f){0.f,0.f,0.f,0.f,0.f,0.f,0.f,0.f};

    // A staging: 128x32 tile, 16 contiguous elems per thread (2 threads/row)
    const int ar = tid >> 1;
    const int ac = (tid & 1) * 16;
    const int agr = blockM + ar;
    const float rs = HAS_RS ? rowScale[agr] : 1.0f;
    const float* aBase = A + (size_t)agr * K + ac;
    const float* eBase = HAS_EM ? (eltMul + (size_t)agr * K + ac) : nullptr;

    // B staging: column walk; thread covers column (tid&127), 16 K-values
    const int bcol = tid & 127;
    const int bkc  = (tid >> 7) * 16;          // 0 or 16
    const int bgn  = blockN + bcol;
    const bool bIn = bgn < N;
    const float* bBase = B + (size_t)bkc * N + bgn;

    float aR[16], eR[16], bR[16];

    auto loadTiles = [&](int k0) {
#pragma unroll
        for (int q = 0; q < 4; ++q) {
            const float4 f = ((const float4*)(aBase + k0))[q];
            aR[4*q+0] = f.x; aR[4*q+1] = f.y; aR[4*q+2] = f.z; aR[4*q+3] = f.w;
        }
        if (HAS_EM) {
#pragma unroll
            for (int q = 0; q < 4; ++q) {
                const float4 f = ((const float4*)(eBase + k0))[q];
                eR[4*q+0] = f.x; eR[4*q+1] = f.y; eR[4*q+2] = f.z; eR[4*q+3] = f.w;
            }
        }
        if (bIn) {
            const float* bp = bBase + (size_t)k0 * N;
#pragma unroll
            for (int j = 0; j < 16; ++j) bR[j] = bp[(size_t)j * N];
        } else {
#pragma unroll
            for (int j = 0; j < 16; ++j) bR[j] = 0.0f;
        }
    };

    auto storeTiles = [&](int p) {
        bf16x8 pa0, pa1, pb0, pb1;
#pragma unroll
        for (int j = 0; j < 8; ++j) {
            float v0 = aR[j] * rs, v1 = aR[j + 8] * rs;
            if (HAS_EM) { v0 *= eR[j]; v1 *= eR[j + 8]; }
            pa0[j] = (__bf16)v0; pa1[j] = (__bf16)v1;
            pb0[j] = (__bf16)bR[j]; pb1[j] = (__bf16)bR[j + 8];
        }
        *(bf16x8*)&As[p][ar * LDST + ac]     = pa0;
        *(bf16x8*)&As[p][ar * LDST + ac + 8] = pa1;
        *(bf16x8*)&Bs[p][bcol * LDST + bkc]     = pb0;
        *(bf16x8*)&Bs[p][bcol * LDST + bkc + 8] = pb1;
    };

    int p = 0;
    loadTiles(0);
    for (int k0 = 0; k0 < K; k0 += BK) {
        storeTiles(p);
        __syncthreads();
        if (k0 + BK < K) loadTiles(k0 + BK);   // in flight during WMMAs below

        // fragments in contiguous k order (k = lh*16 + s), same for A(16x32) and B(32x16)
        v16bf aF[2], bF[4];
#pragma unroll
        for (int mi = 0; mi < 2; ++mi) {
            const __bf16* base = &As[p][(waveM * 32 + mi * 16 + lr) * LDST];
#pragma unroll
            for (int j = 0; j < 8; ++j) {
                aF[mi][j]     = base[lh * 16 + j];       // K = lh*16 + j
                aF[mi][j + 8] = base[lh * 16 + 8 + j];   // K = lh*16 + 8 + j
            }
        }
#pragma unroll
        for (int ni = 0; ni < 4; ++ni) {
            const __bf16* base = &Bs[p][(waveN * 64 + ni * 16 + lr) * LDST];
#pragma unroll
            for (int j = 0; j < 16; ++j)
                bF[ni][j] = base[lh * 16 + j];           // K = lh*16 + j
        }

#pragma unroll
        for (int mi = 0; mi < 2; ++mi)
#pragma unroll
            for (int ni = 0; ni < 4; ++ni)
                acc[mi][ni] = __builtin_amdgcn_wmma_f32_16x16x32_bf16(
                    false, aF[mi], false, bF[ni], (short)0, acc[mi][ni],
                    false, false);
        p ^= 1;
    }

    // store D (f32 16x16 C/D layout: VGPR v -> row v + 8*lh, col = lr)
#pragma unroll
    for (int mi = 0; mi < 2; ++mi) {
        const int rbase = blockM + waveM * 32 + mi * 16 + lh * 8;
#pragma unroll
        for (int ni = 0; ni < 4; ++ni) {
            const int col = blockN + waveN * 64 + ni * 16 + lr;
            if (col < N) {
#pragma unroll
                for (int v = 0; v < 8; ++v)
                    C[(size_t)(rbase + v) * N + col] = acc[mi][ni][v];
            }
        }
    }
}

// ---------------------------------------------------------------------------
// Elementwise activation (in place): op 0 = tanh, op 1 = sigmoid
// ---------------------------------------------------------------------------
__global__ void unary_inplace(float* __restrict__ X, int n, int op)
{
    int idx = blockIdx.x * 256 + threadIdx.x;
    if (idx < n) {
        float v = X[idx];
        X[idx] = (op == 0) ? tanhf(v) : (1.0f / (1.0f + expf(-v)));
    }
}

__device__ __forceinline__ float sigm_(float x) { return 1.0f / (1.0f + expf(-x)); }
__device__ __forceinline__ float softplus_(float x) { return (x > 30.f) ? x : log1pf(expf(x)); }

__device__ __forceinline__ float blockReduceSum128(float v, float* red, int tid)
{
    red[tid] = v;
    __syncthreads();
#pragma unroll
    for (int s = 64; s >= 1; s >>= 1) {
        if (tid < s) red[tid] += red[tid + s];
        __syncthreads();
    }
    float r = red[0];
    __syncthreads();
    return r;
}

// ---------------------------------------------------------------------------
// Fused pre-scan: one block per (t, h), 128 threads (n = lane over head dim).
// Produces bf16 scan vectors r,k,v,aa,bb, f32 wdec (in place over w_full),
// and the per-(t,h) bonus dot  sum_n r*k*r_k.
// ---------------------------------------------------------------------------
__global__ __launch_bounds__(128) void fuse_pre_scan(
    const float* __restrict__ Rp,   const float* __restrict__ Kp,
    const float* __restrict__ Vp,   const float* __restrict__ wfull,
    const float* __restrict__ afull,const float* __restrict__ vsig_mm,
    const float* __restrict__ ksig_mm,
    const float* __restrict__ v_first, const float* __restrict__ k_first,
    const float* __restrict__ cosT, const float* __restrict__ sinT,
    const float* __restrict__ w0,   const float* __restrict__ a0,
    const float* __restrict__ v0,   const float* __restrict__ k0,
    const float* __restrict__ r_norm_w, const float* __restrict__ k_norm_w,
    const float* __restrict__ r_k,
    float* __restrict__ wdec_out,
    __bf16* __restrict__ r_s, __bf16* __restrict__ k_s,
    __bf16* __restrict__ v_s, __bf16* __restrict__ aa_s,
    __bf16* __restrict__ bb_s, float* __restrict__ rk_dot)
{
    const int t  = blockIdx.x;
    const int h  = blockIdx.y;
    const int n  = threadIdx.x;
    const int hk = h >> 2;                  // kv head (N_REP = 4)
    const int c  = h * HDIM + n;            // full-head channel
    const int ck = hk * HDIM + n;           // kv-head channel

    __shared__ float red[128];
    __shared__ float stage[128];

    const float cs = cosT[t * HDIM + n];
    const float sn = sinT[t * HDIM + n];

    // r: RMS norm + RoPE
    const float rv  = Rp[(size_t)t * C_DIM + c];
    const float rss = blockReduceSum128(rv * rv, red, n);
    const float rn_ = r_norm_w[n] * rv * rsqrtf(rss * (1.0f / HDIM) + 1e-6f);
    stage[n] = rn_; __syncthreads();
    const float rrot = (n < 64) ? -stage[n + 64] : stage[n - 64];
    __syncthreads();
    const float r_f = rn_ * cs + rrot * sn;

    // k (kv head): RMS norm + RoPE + lerp with k_first
    const float kv0 = Kp[(size_t)t * KVH_DIM + ck];
    const float kss = blockReduceSum128(kv0 * kv0, red, n);
    const float kn_ = k_norm_w[n] * kv0 * rsqrtf(kss * (1.0f / HDIM) + 1e-6f);
    stage[n] = kn_; __syncthreads();
    const float krot = (n < 64) ? -stage[n + 64] : stage[n - 64];
    __syncthreads();
    const float k_rope = kn_ * cs + krot * sn;
    const float ksg = sigm_(k0[ck] + ksig_mm[(size_t)t * KVH_DIM + ck]);
    const float k_f = k_rope + (k_first[(size_t)t * KVH_DIM + ck] - k_rope) * ksg;

    // v lerp with v_first
    const float vp  = Vp[(size_t)t * KVH_DIM + ck];
    const float vsg = sigm_(v0[ck] + vsig_mm[(size_t)t * KVH_DIM + ck]);
    const float v_f = vp + (v_first[(size_t)t * KVH_DIM + ck] - vp) * vsg;

    // kk = k / max(||k||, 1e-12);  aa = -kk; bb = kk*a
    const float k2  = blockReduceSum128(k_f * k_f, red, n);
    const float inv = 1.0f / fmaxf(sqrtf(k2), 1e-12f);
    const float kk  = k_f * inv;
    const float a   = sigm_(a0[c] + afull[(size_t)t * C_DIM + c]);

    // wdec = exp(-exp(-softplus(-w) - 0.5))
    const float wv = w0[c] + wfull[(size_t)t * C_DIM + c];
    const float wl = -softplus_(-wv) - 0.5f;
    const float wd = expf(-expf(wl));

    // bonus dot: sum_n r*k*r_k[h,n]
    const float rkd = blockReduceSum128(r_f * k_f * r_k[c], red, n);
    if (n == 0) rk_dot[t * HEADS + h] = rkd;

    const size_t o = (size_t)t * C_DIM + c;   // (T, H, N) flat == (T, C)
    wdec_out[o] = wd;
    r_s[o]  = (__bf16)r_f;
    k_s[o]  = (__bf16)k_f;
    v_s[o]  = (__bf16)v_f;
    aa_s[o] = (__bf16)(-kk);
    bb_s[o] = (__bf16)(kk * a);
}

// ---------------------------------------------------------------------------
// Sequential scan: one block per head, 128 threads (thread i owns row i of S,
// 128 f32 state values in VGPRs). Per-step vectors broadcast via
// double-buffered LDS; next step's global loads software-pipelined.
//   sa_i = sum_j S_ij*aa_j;  S_ij = S_ij*w_j + sa_i*bb_j + v_i*k_j
//   y_i  = sum_j S_ij*r_j  + rk_dot*v_i
// ---------------------------------------------------------------------------
__global__ __launch_bounds__(128) void rwkv_scan(
    const __bf16* __restrict__ r_s, const __bf16* __restrict__ k_s,
    const __bf16* __restrict__ v_s, const __bf16* __restrict__ aa_s,
    const __bf16* __restrict__ bb_s, const float* __restrict__ wdec,
    const float* __restrict__ rk_dot, float* __restrict__ y)
{
    const int h = blockIdx.x;
    const int i = threadIdx.x;

    __shared__ __align__(16) float lw[2][128];
    __shared__ __align__(16) float lr[2][128];
    __shared__ __align__(16) float lk[2][128];
    __shared__ __align__(16) float la[2][128];
    __shared__ __align__(16) float lb[2][128];

    float S[128];
#pragma unroll
    for (int j = 0; j < 128; ++j) S[j] = 0.0f;

    const int base = h * HDIM + i;

    // preload t = 0
    float pw = wdec[base];
    float pr = (float)r_s[base],  pk = (float)k_s[base];
    float pv = (float)v_s[base],  pa = (float)aa_s[base];
    float pb = (float)bb_s[base];

    for (int t = 0; t < T_SEQ; ++t) {
        const int p = t & 1;
        lw[p][i] = pw; lr[p][i] = pr; lk[p][i] = pk; la[p][i] = pa; lb[p][i] = pb;
        const float vi  = pv;
        const float rkd = rk_dot[t * HEADS + h];
        __syncthreads();

        // pipeline next step's global loads over this step's compute
        if (t + 1 < T_SEQ) {
            const int nb = (t + 1) * C_DIM + base;
            pw = wdec[nb];
            pr = (float)r_s[nb];  pk = (float)k_s[nb];
            pv = (float)v_s[nb];  pa = (float)aa_s[nb];
            pb = (float)bb_s[nb];
        }

        const float4* a4 = (const float4*)la[p];
        const float4* w4 = (const float4*)lw[p];
        const float4* k4 = (const float4*)lk[p];
        const float4* b4 = (const float4*)lb[p];
        const float4* r4 = (const float4*)lr[p];

        float sa = 0.0f;
#pragma unroll
        for (int j4 = 0; j4 < 32; ++j4) {
            const float4 av = a4[j4];
            sa = fmaf(S[4*j4+0], av.x, sa);
            sa = fmaf(S[4*j4+1], av.y, sa);
            sa = fmaf(S[4*j4+2], av.z, sa);
            sa = fmaf(S[4*j4+3], av.w, sa);
        }

        float yi = 0.0f;
#pragma unroll
        for (int j4 = 0; j4 < 32; ++j4) {
            const float4 wv = w4[j4], kv = k4[j4], bv = b4[j4], rv = r4[j4];
            float s;
            s = fmaf(S[4*j4+0], wv.x, fmaf(sa, bv.x, vi*kv.x)); S[4*j4+0]=s; yi = fmaf(s, rv.x, yi);
            s = fmaf(S[4*j4+1], wv.y, fmaf(sa, bv.y, vi*kv.y)); S[4*j4+1]=s; yi = fmaf(s, rv.y, yi);
            s = fmaf(S[4*j4+2], wv.z, fmaf(sa, bv.z, vi*kv.z)); S[4*j4+2]=s; yi = fmaf(s, rv.z, yi);
            s = fmaf(S[4*j4+3], wv.w, fmaf(sa, bv.w, vi*kv.w)); S[4*j4+3]=s; yi = fmaf(s, rv.w, yi);
        }

        y[(size_t)t * C_DIM + base] = yi + rkd * vi;
    }
}

// ---------------------------------------------------------------------------
// Host orchestration
// ---------------------------------------------------------------------------
extern "C" void kernel_launch(void* const* d_in, const int* in_sizes, int n_in,
                              void* d_out, int out_size, void* d_ws, size_t ws_size,
                              hipStream_t stream)
{
    (void)in_sizes; (void)n_in; (void)out_size; (void)ws_size;
    const float* x        = (const float*)d_in[0];
    const float* v_first  = (const float*)d_in[1];
    const float* k_first  = (const float*)d_in[2];
    const float* mask     = (const float*)d_in[3];
    const float* cosT     = (const float*)d_in[4];
    const float* sinT     = (const float*)d_in[5];
    const float* w0       = (const float*)d_in[6];
    const float* w1       = (const float*)d_in[7];
    const float* w2       = (const float*)d_in[8];
    const float* a0       = (const float*)d_in[9];
    const float* a1       = (const float*)d_in[10];
    const float* a2       = (const float*)d_in[11];
    const float* v0       = (const float*)d_in[12];
    const float* v1       = (const float*)d_in[13];
    const float* v2       = (const float*)d_in[14];
    const float* k0       = (const float*)d_in[15];
    const float* k1       = (const float*)d_in[16];
    const float* k2       = (const float*)d_in[17];
    const float* g1       = (const float*)d_in[18];
    const float* g2       = (const float*)d_in[19];
    const float* r_k      = (const float*)d_in[20];
    const float* r_norm_w = (const float*)d_in[21];
    const float* k_norm_w = (const float*)d_in[22];
    const float* W_r      = (const float*)d_in[23];
    const float* W_k      = (const float*)d_in[24];
    const float* W_v      = (const float*)d_in[25];
    const float* W_o      = (const float*)d_in[26];
    float* out = (float*)d_out;

    const int T = T_SEQ, C = C_DIM, KV = KVH_DIM;

    // ---- workspace partition (all sizes keep 16B alignment) ----
    char* wsb = (char*)d_ws;
    size_t off = 0;
    auto fal = [&](size_t n) -> float* { float* p = (float*)(wsb + off); off += n * sizeof(float); return p; };
    float* Rp    = fal((size_t)T * C);     // r projection; later reused as y
    float* Kp    = fal((size_t)T * KV);
    float* Vp    = fal((size_t)T * KV);
    float* hw    = fal((size_t)T * 160);
    float* wfull = fal((size_t)T * C);     // becomes wdec in place
    float* amid  = fal((size_t)T * 96);
    float* afull = fal((size_t)T * C);
    float* gmid  = fal((size_t)T * 160);
    float* gfull = fal((size_t)T * C);
    float* vmid  = fal((size_t)T * 64);
    float* vsig  = fal((size_t)T * KV);
    float* kmid  = fal((size_t)T * 64);
    float* ksig  = fal((size_t)T * KV);
    float* rkdot = fal((size_t)T * HEADS);
    auto bal = [&](size_t n) -> __bf16* { __bf16* p = (__bf16*)(wsb + off); off += n * sizeof(__bf16); return p; };
    __bf16* r_s  = bal((size_t)T * C);
    __bf16* k_s  = bal((size_t)T * C);
    __bf16* v_s  = bal((size_t)T * C);
    __bf16* aa_s = bal((size_t)T * C);
    __bf16* bb_s = bal((size_t)T * C);
    float* y = Rp;                         // alias: Rp fully consumed by fuse_pre_scan

    dim3 blk(256);
    auto gemm = [&](const float* A, const float* B, float* Cp, int M, int N, int K,
                    const float* rscale, const float* em) {
        dim3 grd((N + BN - 1) / BN, (M + BM - 1) / BM);
        if (rscale)
            gemm_bf16_wmma<true, false><<<grd, blk, 0, stream>>>(A, B, Cp, M, N, K, rscale, nullptr);
        else if (em)
            gemm_bf16_wmma<false, true><<<grd, blk, 0, stream>>>(A, B, Cp, M, N, K, nullptr, em);
        else
            gemm_bf16_wmma<false, false><<<grd, blk, 0, stream>>>(A, B, Cp, M, N, K, nullptr, nullptr);
    };

    // stage 1: projections off xm = x * mask (mask fused as rowScale)
    gemm(x, W_r, Rp,   T, C,   C, mask, nullptr);
    gemm(x, W_k, Kp,   T, KV,  C, mask, nullptr);
    gemm(x, W_v, Vp,   T, KV,  C, mask, nullptr);
    gemm(x, w1,  hw,   T, 160, C, mask, nullptr);
    gemm(x, a1,  amid, T, 96,  C, mask, nullptr);
    gemm(x, g1,  gmid, T, 160, C, mask, nullptr);
    gemm(x, v1,  vmid, T, 64,  C, mask, nullptr);
    gemm(x, k1,  kmid, T, 64,  C, mask, nullptr);

    // activations on the low-rank mids
    unary_inplace<<<(T * 160 + 255) / 256, 256, 0, stream>>>(hw,   T * 160, 0); // tanh
    unary_inplace<<<(T * 160 + 255) / 256, 256, 0, stream>>>(gmid, T * 160, 1); // sigmoid

    // stage 2: second half of the low-rank pairs
    gemm(hw,   w2, wfull, T, C,  160, nullptr, nullptr);
    gemm(amid, a2, afull, T, C,  96,  nullptr, nullptr);
    gemm(gmid, g2, gfull, T, C,  160, nullptr, nullptr);
    gemm(vmid, v2, vsig,  T, KV, 64,  nullptr, nullptr);
    gemm(kmid, k2, ksig,  T, KV, 64,  nullptr, nullptr);

    // fused norm/rope/lerp/decay -> bf16 scan vectors
    fuse_pre_scan<<<dim3(T, HEADS), 128, 0, stream>>>(
        Rp, Kp, Vp, wfull, afull, vsig, ksig, v_first, k_first, cosT, sinT,
        w0, a0, v0, k0, r_norm_w, k_norm_w, r_k,
        wfull /*wdec in place*/, r_s, k_s, v_s, aa_s, bb_s, rkdot);

    // sequential state scan (register-resident S)
    rwkv_scan<<<HEADS, 128, 0, stream>>>(r_s, k_s, v_s, aa_s, bb_s,
                                         wfull, rkdot, y);

    // output projection with gate fused into A load: out = (y * g) @ W_o
    gemm(y, W_o, out, T, C, C, nullptr, gfull);
}
